// _IPEXGatedMLPMOEXPU_70763881169536
// MI455X (gfx1250) — matrix operand order repaired and checked by reading, not compiled, not run on hardware
//
#include <hip/hip_runtime.h>
#include <hip/hip_bf16.h>
#include <math.h>

// Problem sizes (fixed by the reference).
#define E_ 8
#define T_ 1024
#define H_ 1024
#define I_ 1024

// Block tile: 256 threads = 8 wave32; each wave owns one 16x16 subtile.
#define BM 32
#define BN 64
#define BK 32

typedef __attribute__((ext_vector_type(16))) __bf16 bf16x16;
typedef __attribute__((ext_vector_type(8)))  __bf16 bf16x8;
typedef __attribute__((ext_vector_type(8)))  float  f32x8;
typedef __attribute__((ext_vector_type(4)))  unsigned u32x4;
typedef __attribute__((ext_vector_type(8)))  unsigned u32x8;

// Wave-relative LDS byte offset from a __shared__ pointer: generic LDS
// addresses keep the DS offset in the low 32 bits (ISA 10.2 aperture rules).
__device__ __forceinline__ unsigned lds_off(const void* p) {
  return (unsigned)(uintptr_t)p;
}

// Assemble a 16-element bf16 WMMA fragment from two aligned 16B LDS reads.
__device__ __forceinline__ bf16x16 frag_ld(const __bf16* base, int off_lo, int off_hi) {
  bf16x8 lo = *(const bf16x8*)(base + off_lo);
  bf16x8 hi = *(const bf16x8*)(base + off_hi);
  bf16x16 r;
#pragma unroll
  for (int i = 0; i < 8; ++i) { r[i] = lo[i]; r[i + 8] = hi[i]; }
  return r;
}

// Assemble an A fragment from an f32 LDS row, converting f32->bf16 in regs
// (pairs lower to v_cvt_pk_bf16_f32, co-executes with WMMA).
__device__ __forceinline__ bf16x16 frag_cvt(const float* row, int kb) {
  float4 f0 = *(const float4*)(row + kb);
  float4 f1 = *(const float4*)(row + kb + 4);
  float4 f2 = *(const float4*)(row + kb + 16);
  float4 f3 = *(const float4*)(row + kb + 20);
  bf16x16 a;
  a[0] = (__bf16)f0.x; a[1] = (__bf16)f0.y; a[2]  = (__bf16)f0.z; a[3]  = (__bf16)f0.w;
  a[4] = (__bf16)f1.x; a[5] = (__bf16)f1.y; a[6]  = (__bf16)f1.z; a[7]  = (__bf16)f1.w;
  a[8] = (__bf16)f2.x; a[9] = (__bf16)f2.y; a[10] = (__bf16)f2.z; a[11] = (__bf16)f2.w;
  a[12] = (__bf16)f3.x; a[13] = (__bf16)f3.y; a[14] = (__bf16)f3.z; a[15] = (__bf16)f3.w;
  return a;
}

// ---------------------------------------------------------------- init -----
__global__ void moe_init(float* __restrict__ out, int* __restrict__ cnt) {
  int i = blockIdx.x * 256 + threadIdx.x;
  if (i < T_ * H_) out[i] = 0.0f;
  if (i < E_) cnt[i] = 0;
}

// --------------------------------------------------------------- route -----
__global__ void moe_route(const float* __restrict__ logits, int* __restrict__ cnt,
                          int* __restrict__ tokid, float* __restrict__ gw) {
  int t = blockIdx.x * 256 + threadIdx.x;
  if (t >= T_) return;
  float p[E_];
  float mx = -3.402823466e38f;
#pragma unroll
  for (int e = 0; e < E_; ++e) { p[e] = logits[t * E_ + e]; mx = fmaxf(mx, p[e]); }
  float s = 0.f;
#pragma unroll
  for (int e = 0; e < E_; ++e) { p[e] = __expf(p[e] - mx); s += p[e]; }
  float inv = 1.f / s;
#pragma unroll
  for (int e = 0; e < E_; ++e) p[e] *= inv;
  int i0 = 0;
#pragma unroll
  for (int e = 1; e < E_; ++e) if (p[e] > p[i0]) i0 = e;
  int i1 = (i0 == 0) ? 1 : 0;
#pragma unroll
  for (int e = 0; e < E_; ++e) if (e != i0 && p[e] > p[i1]) i1 = e;
  float denom = p[i0] + p[i1];
  float w0 = p[i0] / denom, w1 = p[i1] / denom;
  int s0 = atomicAdd(&cnt[i0], 1); tokid[i0 * T_ + s0] = t; gw[i0 * T_ + s0] = w0;
  int s1 = atomicAdd(&cnt[i1], 1); tokid[i1 * T_ + s1] = t; gw[i1 * T_ + s1] = w1;
}

// --------------------------------------------------------------- gemm1 -----
// act = silu(x@W13g) * (x@W13u).  A tile (gathered token rows, f32) staged by
// the async-DMA engine (GLOBAL_LOAD_ASYNC_TO_LDS_B128, per-lane addresses);
// B tiles transposed+converted by the ALUs in parallel.
__global__ __launch_bounds__(256)
void moe_gemm1(const float* __restrict__ X, const float* __restrict__ W13,
               const int* __restrict__ cnt, const int* __restrict__ tokid,
               __bf16* __restrict__ act) {
  const int e = blockIdx.z;
  const int n0 = blockIdx.x * BN;
  const int mbase = blockIdx.y * BM;
  const int c = cnt[e];
  if (mbase >= c) return;  // uniform: whole block exits

  __shared__ __align__(16) float  Asf[BM][BK];  // f32 A tile (async-filled)
  __shared__ __align__(16) __bf16 Bg[BN][BK];   // transposed: [n][k]
  __shared__ __align__(16) __bf16 Bu[BN][BK];

  const int tid = threadIdx.x;
  const int lane = tid & 31;
  const int wv = tid >> 5;
  const int mi = wv >> 2;   // 0..1
  const int ni = wv & 3;    // 0..3

  const int am = tid >> 3;            // 0..31 : A row
  const int ak = (tid & 7) << 2;      // 0,4..28 : A k-quad
  const int bf = tid & 63;            // 0..63 : B column (n)
  const int bh = tid >> 6;            // 0..3  : B row (k) phase

  const int slot = mbase + am;
  const int tok = (slot < c) ? tokid[e * T_ + slot] : 0;
  const float* xrow = X + (size_t)tok * H_;
  const float* w13e = W13 + (size_t)e * H_ * (2 * I_);
  const unsigned lds_a = lds_off(&Asf[am][ak]);

  f32x8 cg = {}; f32x8 cu = {};

  for (int k0 = 0; k0 < H_; k0 += BK) {
    // A tile: async DMA straight into LDS (gather, 16B per lane).
    {
      unsigned long long ga = (unsigned long long)(uintptr_t)(xrow + k0 + ak);
      asm volatile("global_load_async_to_lds_b128 %0, %1, off"
                   :: "v"(lds_a), "v"(ga) : "memory");
    }
    // B tiles: load f32 rows (coalesced along f), convert, store transposed.
#pragma unroll
    for (int s = 0; s < BK; s += 4) {
      const float* rowp = w13e + (size_t)(k0 + s + bh) * (2 * I_);
      Bg[bf][s + bh] = (__bf16)rowp[n0 + bf];
      Bu[bf][s + bh] = (__bf16)rowp[I_ + n0 + bf];
    }
    if (k0 + BK < H_)  // global_prefetch_b8 for the next stage's weights
      __builtin_prefetch(w13e + (size_t)(k0 + BK + bh) * (2 * I_) + n0 + bf, 0, 1);
    asm volatile("s_wait_asynccnt 0x0" ::: "memory");
    __syncthreads();

    const int arow = mi * 16 + (lane & 15);
    const int akb = (lane < 16) ? 0 : 8;       // A: K {0..7,16..23} / {8..15,24..31}
    bf16x16 a = frag_cvt(&Asf[arow][0], akb);
    const int bcol = ni * 16 + (lane & 15);
    const int bkb = (lane < 16) ? 0 : 8;       // B: K {0..7,16..23} / {8..15,24..31}
    bf16x16 bg = frag_ld(&Bg[bcol][0], bkb, bkb + 16);
    bf16x16 bu = frag_ld(&Bu[bcol][0], bkb, bkb + 16);

    cg = __builtin_amdgcn_wmma_f32_16x16x32_bf16(false, a, false, bg, (short)0, cg, false, false);
    cu = __builtin_amdgcn_wmma_f32_16x16x32_bf16(false, a, false, bu, (short)0, cu, false, false);
    __syncthreads();
  }

  const int colg = n0 + ni * 16 + (lane & 15);
#pragma unroll
  for (int r = 0; r < 8; ++r) {
    int sm = mbase + mi * 16 + r + ((lane < 16) ? 0 : 8);  // C: VGPR r <-> M=r/r+8
    if (sm < c) {
      float g = cg[r], u = cu[r];
      float a = g / (1.f + __expf(-g)) * u;  // silu(g) * u
      act[((size_t)e * T_ + sm) * I_ + colg] = (__bf16)a;
    }
  }
}

// --------------------------------------------------------------- gemm2 -----
// out[tok] += w * (act @ W2).  A tile (bf16, uniform base + row stride) is a
// textbook 2D TDM tile: one wave issues tensor_load_to_lds per K stage.
__global__ __launch_bounds__(256)
void moe_gemm2(const __bf16* __restrict__ act, const float* __restrict__ W2,
               const int* __restrict__ cnt, const int* __restrict__ tokid,
               const float* __restrict__ gw, float* __restrict__ out) {
  const int e = blockIdx.z;
  const int n0 = blockIdx.x * BN;
  const int mbase = blockIdx.y * BM;
  const int c = cnt[e];
  if (mbase >= c) return;

  __shared__ __align__(16) __bf16 As[BM][BK];   // bf16 A tile (TDM-filled)
  __shared__ __align__(16) __bf16 Bs[BN][BK];   // transposed: [n][k]

  const int tid = threadIdx.x;
  const int lane = tid & 31;
  const int wv = tid >> 5;
  const int mi = wv >> 2;
  const int ni = wv & 3;

  const int bf = tid & 63;
  const int bh = tid >> 6;

  const float* w2e = W2 + (size_t)e * I_ * H_;
  // bf16 act rows for this tile: base + k0*2 bytes each stage, stride I_*2.
  const unsigned long long abase =
      (unsigned long long)(uintptr_t)(act + ((size_t)e * T_ + mbase) * I_);

  // Tensor DMA descriptor group 1 (invariant): data_size=2B, tensor_dim0=I_,
  // tensor_dim1=32 rows, tile 32x32, dim0 stride = I_ elements.
  const u32x8 g1 = {0x00010000u,                 // wg_mask=0, data_size=1(2B)
                    (unsigned)(I_ & 0xFFFF) << 16,   // tensor_dim0 lo16
                    ((unsigned)(I_ >> 16)) | (32u << 16),  // dim0 hi16 | tensor_dim1 lo16
                    (32u << 16),                 // tensor_dim1 hi16(0) | tile_dim0=32
                    32u,                         // tile_dim1=32, tile_dim2=0
                    (unsigned)I_,                // tensor_dim0_stride lo32
                    0u, 0u};                     // stride hi | dim1_stride (unused)
  const u32x4 gz = {0u, 0u, 0u, 0u};             // groups 2/3: dims beyond 2D unused
  const unsigned lds_a = lds_off(&As[0][0]);

  f32x8 cc = {};

  for (int k0 = 0; k0 < I_; k0 += BK) {
    if (wv == 0) {  // uniform branch: wave 0 drives the TDM engine
      unsigned long long ga = abase + (unsigned long long)(k0 * 2);
      u32x4 g0 = {1u,                                  // count=1 valid descriptor
                  lds_a,                               // lds_addr
                  (unsigned)(ga & 0xFFFFFFFFu),        // global_addr[31:0]
                  (unsigned)((ga >> 32) & 0x01FFFFFFu) | 0x80000000u};  // addr[56:32] | type=2
      asm volatile("tensor_load_to_lds %0, %1, %2, %3"
                   :: "s"(g0), "s"(g1), "s"(gz), "s"(gz) : "memory");
    }
    // B tile: f32 W2 rows -> bf16, transposed (overlaps the TDM transfer).
#pragma unroll
    for (int s = 0; s < BK; s += 4) {
      const float* rowp = w2e + (size_t)(k0 + s + bh) * H_;
      Bs[bf][s + bh] = (__bf16)rowp[n0 + bf];
    }
    if (k0 + BK < I_)
      __builtin_prefetch(w2e + (size_t)(k0 + BK + bh) * H_ + n0 + bf, 0, 1);
    if (wv == 0) __builtin_amdgcn_s_wait_tensorcnt(0);
    __syncthreads();

    const int arw = mi * 16 + (lane & 15);
    const int akb = (lane < 16) ? 0 : 8;
    bf16x16 a = frag_ld(&As[arw][0], akb, akb + 16);
    const int bcol = ni * 16 + (lane & 15);
    const int bkb = (lane < 16) ? 0 : 8;
    bf16x16 b = frag_ld(&Bs[bcol][0], bkb, bkb + 16);

    cc = __builtin_amdgcn_wmma_f32_16x16x32_bf16(false, a, false, b, (short)0, cc, false, false);
    __syncthreads();
  }

  const int colh = n0 + ni * 16 + (lane & 15);
#pragma unroll
  for (int r = 0; r < 8; ++r) {
    int sm = mbase + mi * 16 + r + ((lane < 16) ? 0 : 8);
    if (sm < c) {
      int t = tokid[e * T_ + sm];
      float wgt = gw[e * T_ + sm];
      unsafeAtomicAdd(out + (size_t)t * H_ + colh, wgt * cc[r]);
    }
  }
}

// -------------------------------------------------------------- launch -----
extern "C" void kernel_launch(void* const* d_in, const int* in_sizes, int n_in,
                              void* d_out, int out_size, void* d_ws, size_t ws_size,
                              hipStream_t stream) {
  const float* X   = (const float*)d_in[0];  // [T,H]
  const float* RL  = (const float*)d_in[1];  // [T,E]
  const float* W13 = (const float*)d_in[2];  // [E,H,2I]
  const float* W2  = (const float*)d_in[3];  // [E,I,H]
  // d_in[4] = top_k (=2, baked into routing)
  float* out = (float*)d_out;                // [T,H]

  // Workspace (~16.1 MB): counters | E*T token ids | E*T weights | bf16 act.
  int*    cnt   = (int*)d_ws;
  int*    tokid = cnt + 16;
  float*  gw    = (float*)(tokid + E_ * T_);
  __bf16* act   = (__bf16*)(gw + E_ * T_);

  moe_init <<<(T_ * H_ + 255) / 256, 256, 0, stream>>>(out, cnt);
  moe_route<<<(T_ + 255) / 256,      256, 0, stream>>>(RL, cnt, tokid, gw);
  moe_gemm1<<<dim3(I_ / BN, T_ / BM, E_), 256, 0, stream>>>(X, W13, cnt, tokid, act);
  moe_gemm2<<<dim3(H_ / BN, T_ / BM, E_), 256, 0, stream>>>(act, W2, cnt, tokid, gw, out);
}
